// GraphVectorQuantizer_27650999452273
// MI455X (gfx1250) — matrix operand order repaired and checked by reading, not compiled, not run on hardware
//
#include <hip/hip_runtime.h>
#include <hip/hip_bf16.h>
#include <stdint.h>

#if __has_builtin(__builtin_amdgcn_global_load_async_to_lds_b128)
#define ASYNC_COPY 1
#endif

typedef __attribute__((ext_vector_type(16))) __bf16 v16bf;
typedef __attribute__((ext_vector_type(8)))  __bf16 v8bf;
typedef __attribute__((ext_vector_type(8)))  float  v8f;

typedef int v4i_vec __attribute__((vector_size(16)));
typedef __attribute__((address_space(1))) v4i_vec* as1_v4i_ptr;
typedef __attribute__((address_space(3))) v4i_vec* as3_v4i_ptr;

#define B_ROWS 16384
#define DHALF  512
#define TWO_D  1024
#define KCB    4096

#define BM 64
#define BN 32
#define BK 64
#define LDA 72   // padded LDS row stride in bf16 elems (144 B, 16B-aligned, conflict-spread)
#define NSTAGE ((KCB / BN) * (TWO_D / BK))   // 2048 pipeline stages

__device__ __forceinline__ v8f wmma_bf16(v16bf a, v16bf b, v8f c) {
  return __builtin_amdgcn_wmma_f32_16x16x32_bf16(false, a, false, b, (short)0, c, false, false);
}

// A fragment (16x32 bf16): lane m=l&15, hb=l>>4.
// elems 0..7  -> k = 8*hb + (0..7); elems 8..15 -> k = 16+8*hb + (0..7)
__device__ __forceinline__ v16bf load_fragA(const __bf16* base, int hb) {
  v8bf x = *(const v8bf*)(base + 8 * hb);
  v8bf y = *(const v8bf*)(base + 16 + 8 * hb);
  return __builtin_shufflevector(x, y, 0,1,2,3,4,5,6,7,8,9,10,11,12,13,14,15);
}

// B fragment (32x16 bf16): lane n=l&15, hb=l>>4; elems 0..7 -> k = 8*hb + (0..7); elems 8..15 -> k = 16+8*hb + (0..7)
__device__ __forceinline__ v16bf load_fragB(const __bf16* base) {
  v8bf x = *(const v8bf*)(base);
  v8bf y = *(const v8bf*)(base + 16);
  return __builtin_shufflevector(x, y, 0,1,2,3,4,5,6,7,8,9,10,11,12,13,14,15);
}

// 16B memory -> LDS copy: async DMA path on CDNA5, VGPR round-trip fallback.
__device__ __forceinline__ void copy16_to_lds(const void* g, void* l) {
#ifdef ASYNC_COPY
  __builtin_amdgcn_global_load_async_to_lds_b128(
      (as1_v4i_ptr)(v4i_vec*)g,
      (as3_v4i_ptr)(v4i_vec*)l, 0, 0);
#else
  *(uint4*)l = *(const uint4*)g;
#endif
}

__device__ __forceinline__ void wait_async_done() {
#ifdef ASYNC_COPY
#if __has_builtin(__builtin_amdgcn_s_wait_asynccnt)
  __builtin_amdgcn_s_wait_asynccnt(0);
#else
  asm volatile("s_wait_asynccnt 0x0" ::: "memory");
#endif
#endif
}

// ---------------- prep kernels ----------------

__global__ __launch_bounds__(256) void vq_prep_z(const float* __restrict__ zr,
    const float* __restrict__ zi, const float* __restrict__ noise,
    __bf16* __restrict__ zh, __bf16* __restrict__ zl)
{
  size_t i = (size_t)blockIdx.x * 256 + threadIdx.x;     // over B*2D
  int b = (int)(i >> 10);
  int j = (int)(i & 1023);
  float v = ((j < DHALF) ? zr[(size_t)b * DHALF + j]
                         : zi[(size_t)b * DHALF + (j - DHALF)]) + noise[i];
  __bf16 h = (__bf16)v;
  zh[i] = h;
  zl[i] = (__bf16)(v - (float)h);
}

__global__ __launch_bounds__(256) void vq_prep_cb(const float* __restrict__ cb,
    __bf16* __restrict__ cbh, __bf16* __restrict__ cbl, float* __restrict__ cbsq)
{
  int k = blockIdx.x;
  int t = threadIdx.x;
  float s = 0.0f;
  #pragma unroll
  for (int e = 0; e < 4; e++) {
    int j = t + e * 256;
    float v = cb[(size_t)k * TWO_D + j];
    __bf16 h = (__bf16)v;
    cbh[(size_t)k * TWO_D + j] = h;
    cbl[(size_t)k * TWO_D + j] = (__bf16)(v - (float)h);
    s += v * v;
  }
  __shared__ float red[256];
  red[t] = s;
  __syncthreads();
  for (int off = 128; off > 0; off >>= 1) {
    if (t < off) red[t] += red[t + off];
    __syncthreads();
  }
  if (t == 0) cbsq[k] = red[0];
}

// ---------------- fused distance GEMM + bias + argmin ----------------

__global__ __launch_bounds__(256) void vq_main(
    const __bf16* __restrict__ zh, const __bf16* __restrict__ zl,
    const __bf16* __restrict__ cbh, const __bf16* __restrict__ cbl,
    const float* __restrict__ cbsq, const int* __restrict__ prev,
    const float* __restrict__ adj,
    float* __restrict__ minval, int* __restrict__ minidx)
{
  __shared__ __bf16 sA[2][2][BM][LDA];   // [buf][hi/lo][row][k]  (36,864 B)
  __shared__ __bf16 sB[2][2][BN][LDA];   // [buf][hi/lo][n][k]    (18,432 B)
  __shared__ int   sPrev[BM];
  __shared__ float sRedV[2][BM];
  __shared__ int   sRedI[2][BM];

  const int tid  = threadIdx.x;
  const int b0   = blockIdx.x * BM;
  const int wave = tid >> 5;
  const int lane = tid & 31;
  const int wm   = wave >> 1;   // 0..3 : 16-row slice
  const int wn   = wave & 1;    // 0..1 : 16-col slice
  const int nl   = lane & 15;
  const int hb   = lane >> 4;

  if (tid < BM) sPrev[tid] = prev[b0 + tid];

  // ---- fixed per-thread staging assignment (fully unrolled, no runtime selects) ----
  // A: 1024 16B-chunks (hi 512 + lo 512); thread handles c = tid + 256*q, q=0..3
  const char* gA[4]; uint32_t lAoff[4];
  #pragma unroll
  for (int q = 0; q < 4; q++) {
    int c = tid + 256 * q;
    int mtx = c >> 9;             // 0=hi, 1=lo
    int r   = (c >> 3) & 63;
    int o   = c & 7;
    gA[q]    = (const char*)((mtx ? zl : zh) + (size_t)(b0 + r) * TWO_D + o * 8);
    lAoff[q] = (uint32_t)(((mtx * BM + r) * LDA + o * 8) * sizeof(__bf16));
  }
  // B: 512 16B-chunks; thread handles c = tid + 256*q, q=0..1
  const char* gB[2]; uint32_t lBoff[2];
  #pragma unroll
  for (int q = 0; q < 2; q++) {
    int c = tid + 256 * q;
    int mtx = c >> 8;
    int r   = (c >> 3) & 31;
    int o   = c & 7;
    gB[q]    = (const char*)((mtx ? cbl : cbh) + (size_t)r * TWO_D + o * 8);
    lBoff[q] = (uint32_t)(((mtx * BN + r) * LDA + o * 8) * sizeof(__bf16));
  }
  const int halfA = (int)(sizeof(sA) / 2);   // 18,432 B per buffer
  const int halfB = (int)(sizeof(sB) / 2);   // 9,216 B per buffer

  auto issue_stage = [&](int s) {
    int ko = (s & 15) << 7;       // k-chunk byte offset (64 bf16 = 128 B)
    int no = (s >> 4) << 16;      // n-block byte offset (32 rows * 2048 B)
    char* la = (char*)&sA[0][0][0][0] + (s & 1) * halfA;
    char* lb = (char*)&sB[0][0][0][0] + (s & 1) * halfB;
    #pragma unroll
    for (int q = 0; q < 4; q++) copy16_to_lds(gA[q] + ko, la + lAoff[q]);
    #pragma unroll
    for (int q = 0; q < 2; q++) copy16_to_lds(gB[q] + no + ko, lb + lBoff[q]);
  };

  float rmin[8];
  int   ridx[8];
  #pragma unroll
  for (int j = 0; j < 8; j++) { rmin[j] = 3.4e38f; ridx[j] = 0x7fffffff; }

  issue_stage(0);                 // prologue: fill buffer 0
  int stage = 0;

  for (int n0 = 0; n0 < KCB; n0 += BN) {
    // three independent accumulators -> 3 parallel WMMA dependency chains
    v8f acc0, acc1, acc2;
    #pragma unroll
    for (int q = 0; q < 8; q++) { acc0[q] = 0.0f; acc1[q] = 0.0f; acc2[q] = 0.0f; }

    for (int k0 = 0; k0 < TWO_D; k0 += BK) {
      wait_async_done();                 // my stage-`stage` chunks have landed
      __syncthreads();                   // everyone's landed; everyone's done with buf^1
      if (stage + 1 < NSTAGE) issue_stage(stage + 1);   // prefetch into other buffer

      const int buf = stage & 1;
      #pragma unroll
      for (int ks = 0; ks < 2; ks++) {
        v16bf ah = load_fragA(&sA[buf][0][wm * 16 + nl][ks * 32], hb);
        v16bf al = load_fragA(&sA[buf][1][wm * 16 + nl][ks * 32], hb);
        v16bf bh = load_fragB(&sB[buf][0][wn * 16 + nl][ks * 32 + 8 * hb]);
        v16bf bl = load_fragB(&sB[buf][1][wn * 16 + nl][ks * 32 + 8 * hb]);
        // s = zh*ch + zh*cl + zl*ch (bf16 hi/lo split, ~fp32 accuracy); independent chains
        acc0 = wmma_bf16(ah, bh, acc0);
        acc1 = wmma_bf16(ah, bl, acc1);
        acc2 = wmma_bf16(al, bh, acc2);
      }
      stage++;
    }

    // tile epilogue: d' = |c|^2 - 2 s - bias (row term |z|^2 deferred; constant per row)
    int ng  = n0 + wn * 16 + nl;
    float cn = cbsq[ng];
    #pragma unroll
    for (int j = 0; j < 8; j++) {
      int row = wm * 16 + j + 8 * hb;               // C layout: vgpr j -> M=j (+8 hi half)
      float g    = adj[(size_t)sPrev[row] * KCB + ng];
      float bias = fminf(0.8f / (1.0f + __expf(-g)), 0.5f);
      float sdot = acc0[j] + acc1[j] + acc2[j];
      float dv   = cn - 2.0f * sdot - bias;
      if (dv < rmin[j]) { rmin[j] = dv; ridx[j] = ng; }   // strict < keeps lowest index
    }
  }

  // cross-lane argmin reduction over the 16 N-lanes of each half-wave
  #pragma unroll
  for (int j = 0; j < 8; j++) {
    float v = rmin[j];
    int   id = ridx[j];
    for (int off = 8; off >= 1; off >>= 1) {
      float ov = __shfl_xor(v, off, 32);
      int   oi = __shfl_xor(id, off, 32);
      if (ov < v || (ov == v && oi < id)) { v = ov; id = oi; }
    }
    if (nl == 0) {
      int row = wm * 16 + j + 8 * hb;
      sRedV[wn][row] = v;
      sRedI[wn][row] = id;
    }
  }
  __syncthreads();
  if (tid < BM) {
    float v0 = sRedV[0][tid], v1 = sRedV[1][tid];
    int   i0 = sRedI[0][tid], i1 = sRedI[1][tid];
    bool take1 = (v1 < v0) || (v1 == v0 && i1 < i0);
    minval[b0 + tid] = take1 ? v1 : v0;
    minidx[b0 + tid] = take1 ? i1 : i0;
  }
}

// ---------------- per-row epilogue ----------------

__global__ __launch_bounds__(256) void vq_epilogue(
    const float* __restrict__ zr, const float* __restrict__ zi,
    const float* __restrict__ noise, const float* __restrict__ cb,
    const float* __restrict__ w_score, const float* __restrict__ b_score,
    const float* __restrict__ w_conf, const float* __restrict__ b_conf,
    const float* __restrict__ minval, const int* __restrict__ minidx,
    float* __restrict__ out_real, float* __restrict__ out_imag,
    float* __restrict__ out_score, float* __restrict__ out_conf,
    float* __restrict__ out_idx, float* __restrict__ loss_acc)
{
  int b = blockIdx.x, t = threadIdx.x;
  int idx = minidx[b];
  float ss = 0.0f, sc = 0.0f, sd = 0.0f, sz = 0.0f;
  #pragma unroll
  for (int e = 0; e < 4; e++) {
    int j = t + e * 256;
    float zv = ((j < DHALF) ? zr[(size_t)b * DHALF + j]
                            : zi[(size_t)b * DHALF + j - DHALF]) + noise[(size_t)b * TWO_D + j];
    float q = cb[(size_t)idx * TWO_D + j];
    if (j < DHALF) out_real[(size_t)b * DHALF + j] = q;
    else           out_imag[(size_t)b * DHALF + j - DHALF] = q;
    ss += q * w_score[j];
    sc += q * w_conf[j];
    float df = q - zv;
    sd += df * df;
    sz += zv * zv;
  }
  __shared__ float red[4][256];
  red[0][t] = ss; red[1][t] = sc; red[2][t] = sd; red[3][t] = sz;
  __syncthreads();
  for (int off = 128; off > 0; off >>= 1) {
    if (t < off) {
      red[0][t] += red[0][t + off];
      red[1][t] += red[1][t + off];
      red[2][t] += red[2][t + off];
      red[3][t] += red[3][t + off];
    }
    __syncthreads();
  }
  if (t == 0) {
    float dmin = minval[b] + red[3][0];           // add deferred |z|^2
    float dist_score = -dmin;
    out_score[b] = red[0][0] + b_score[0] + 0.1f * dist_score;
    float cv = red[1][0] + b_conf[0];
    out_conf[b] = 1.0f / (1.0f + __expf(-cv));
    out_idx[b] = (float)idx;
    atomicAdd(loss_acc, red[2][0]);
  }
}

__global__ void vq_zero(float* p) { *p = 0.0f; }

__global__ void vq_finalize(const float* __restrict__ loss_acc, float* __restrict__ out_loss) {
  // loss_vq + 0.01*loss_commit = 1.01 * mean((z_q - z)^2)
  *out_loss = 1.01f * (*loss_acc) / ((float)B_ROWS * (float)TWO_D);
}

// ---------------- launcher ----------------

extern "C" void kernel_launch(void* const* d_in, const int* in_sizes, int n_in,
                              void* d_out, int out_size, void* d_ws, size_t ws_size,
                              hipStream_t stream)
{
  const float* z_real  = (const float*)d_in[0];
  const float* z_imag  = (const float*)d_in[1];
  const float* noise   = (const float*)d_in[2];
  const int*   prev    = (const int*)  d_in[3];
  const float* cb      = (const float*)d_in[4];
  const float* adj     = (const float*)d_in[5];
  const float* w_score = (const float*)d_in[6];
  const float* b_score = (const float*)d_in[7];
  const float* w_conf  = (const float*)d_in[8];
  const float* b_conf  = (const float*)d_in[9];

  char* ws = (char*)d_ws;
  __bf16* zh     = (__bf16*)(ws);                      // 32 MB
  __bf16* zl     = (__bf16*)(ws + 33554432);           // 32 MB
  __bf16* cbh    = (__bf16*)(ws + 67108864);           // 8 MB
  __bf16* cbl    = (__bf16*)(ws + 75497472);           // 8 MB
  float*  cbsq   = (float*) (ws + 83886080);           // 16 KB
  float*  minval = (float*) (ws + 83902464);           // 64 KB
  int*    minidx = (int*)   (ws + 83968000);           // 64 KB
  float*  loss   = (float*) (ws + 84033536);           // 4 B

  float* out       = (float*)d_out;
  float* out_real  = out;                                  // B*D
  float* out_imag  = out + (size_t)B_ROWS * DHALF;         // B*D
  float* out_score = out + 2 * (size_t)B_ROWS * DHALF;     // B
  float* out_conf  = out_score + B_ROWS;                   // B
  float* out_loss  = out_conf + B_ROWS;                    // 1
  float* out_idx   = out_loss + 1;                         // B

  vq_zero<<<1, 1, 0, stream>>>(loss);
  vq_prep_z<<<(B_ROWS * TWO_D) / 256, 256, 0, stream>>>(z_real, z_imag, noise, zh, zl);
  vq_prep_cb<<<KCB, 256, 0, stream>>>(cb, cbh, cbl, cbsq);
  vq_main<<<B_ROWS / BM, 256, 0, stream>>>(zh, zl, cbh, cbl, cbsq, prev, adj, minval, minidx);
  vq_epilogue<<<B_ROWS, 256, 0, stream>>>(z_real, z_imag, noise, cb, w_score, b_score,
                                          w_conf, b_conf, minval, minidx,
                                          out_real, out_imag, out_score, out_conf,
                                          out_idx, loss);
  vq_finalize<<<1, 1, 0, stream>>>(loss, out_loss);
}
